// FastMHA_88622355186165
// MI455X (gfx1250) — compile-verified, matrix operand order checked by reading, not run on hardware
//
#include <hip/hip_runtime.h>
#include <hip/hip_bf16.h>

// ---------------------------------------------------------------------------
// DeBERTa disentangled attention for MI455X (gfx1250, wave32, WMMA bf16).
// - All GEMMs: v_wmma_f32_16x16x32_bf16, LDS-tiled.
// - Global->LDS staging uses async copies (global_load_async_to_lds_b128,
//   ASYNCcnt) with double-buffered LDS to overlap HBM traffic with WMMA.
// - Flash-style attention: never materializes [B,H,L,L]; one pass over
//   pos_indices serving both p2c and c2p gathers.
// ---------------------------------------------------------------------------

#define B_  2
#define L_  1024
#define D_  1024
#define H_  16
#define DK_ 64
#define R_  130
#define SCALE_ 0.07216878364870323f   // 1/sqrt(DK*3)

typedef unsigned short u16;
typedef unsigned int   u32;
typedef unsigned long long u64;
typedef __attribute__((ext_vector_type(16))) __bf16 v16bf;
typedef __attribute__((ext_vector_type(8)))  float  v8f;

union FragBF { v16bf bf; u32 u[8]; };

// k position (even) of the bf16 pair held in element-pair v of a 16-bit
// A/B WMMA operand, for lane group g = lane>>4 (ISA 7.12.2 layout tables).
__device__ __forceinline__ int kpos(int v, int g) {
    return (v < 4) ? (2 * v + 8 * g) : (16 + 2 * (v - 4) + 8 * g);
}

__device__ __forceinline__ u16 f2bf(float f) {
    u32 u = __float_as_uint(f);
    u = (u + 0x7FFFu + ((u >> 16) & 1u)) >> 16;   // RNE
    return (u16)u;
}

// Async 16B global -> LDS copy, tracked with ASYNCcnt (CDNA5 path).
__device__ __forceinline__ void async_load_b128(void* lds_ptr, const void* gptr) {
    const u32 lo = (u32)(size_t)lds_ptr;            // flat LDS addr, low 32 bits
    asm volatile("global_load_async_to_lds_b128 %0, %1, off"
                 :: "v"(lo), "v"((u64)(size_t)gptr)
                 : "memory");
}
#define S_WAIT_ASYNC(n) asm volatile("s_wait_asynccnt " #n ::: "memory")

__device__ __forceinline__ float half16_max(float x) {
    #pragma unroll
    for (int m = 1; m <= 8; m <<= 1) x = fmaxf(x, __shfl_xor(x, m, 32));
    return x;  // reduced across each 16-lane half (rows live per-half)
}
__device__ __forceinline__ float half16_sum(float x) {
    #pragma unroll
    for (int m = 1; m <= 8; m <<= 1) x += __shfl_xor(x, m, 32);
    return x;
}

// ---------------------------------------------------------------------------
// fp32 -> bf16 conversion (optionally pre-scaled)
// ---------------------------------------------------------------------------
__global__ void k_cvt_bf16(const float* __restrict__ in, u16* __restrict__ out,
                           int n, float scale) {
    int i = blockIdx.x * 256 + threadIdx.x;
    if (i < n) out[i] = f2bf(in[i] * scale);
}

// ---------------------------------------------------------------------------
// 128x128-tile bf16 GEMM core with double-buffered async LDS staging.
// C = A[M,K] * W[N,K]^T, K blocked by 32. 256 threads = 8 waves.
// ldsA/ldsB each hold two 128x32 bf16 buffers (2*8KB each).
// ---------------------------------------------------------------------------
__device__ __forceinline__ void gemm_core_128x128(
    const u16* __restrict__ A, const u16* __restrict__ W,
    int m0, int n0, int K, u16* ldsA, u16* ldsB, v8f acc[8], int tid)
{
    const int wave = tid >> 5, lane = tid & 31;
    const int g = lane >> 4, nl = lane & 15;

    auto issue = [&](int k0, int buf) {   // 4 async instrs per wave
        u16* la = ldsA + buf * 4096;
        u16* lb = ldsB + buf * 4096;
        for (int s = tid; s < 512; s += 256) {
            const int row = s >> 2, c = s & 3;
            async_load_b128(la + s * 8, A + (size_t)(m0 + row) * K + k0 + c * 8);
            async_load_b128(lb + s * 8, W + (size_t)(n0 + row) * K + k0 + c * 8);
        }
    };

    issue(0, 0);
    for (int k0 = 0; k0 < K; k0 += 32) {
        const int cur = (k0 >> 5) & 1;
        if (k0 + 32 < K) {
            issue(k0 + 32, cur ^ 1);   // stream next tile while computing this one
            S_WAIT_ASYNC(4);           // current tile landed; next may stay in flight
        } else {
            S_WAIT_ASYNC(0);
        }
        __syncthreads();

        const u16* la = ldsA + cur * 4096;
        const u16* lb = ldsB + cur * 4096;
        FragBF a;
        const int ml = (wave << 4) + nl;
        #pragma unroll
        for (int v = 0; v < 8; ++v)
            a.u[v] = *reinterpret_cast<const u32*>(la + ml * 32 + kpos(v, g));
        #pragma unroll
        for (int nt = 0; nt < 8; ++nt) {
            FragBF b;
            const int nn = (nt << 4) + nl;
            #pragma unroll
            for (int v = 0; v < 8; ++v)
                b.u[v] = *reinterpret_cast<const u32*>(lb + nn * 32 + kpos(v, g));
            acc[nt] = __builtin_amdgcn_wmma_f32_16x16x32_bf16(
                false, a.bf, false, b.bf, (short)0, acc[nt], false, false);
        }
        __syncthreads();   // protects buffer 'cur' before it is re-staged
    }
}

// ---------------------------------------------------------------------------
// QKV projection: q/k/v = hidden @ W^T + b, written bf16 in [B,H,L,DK].
// grid (16, 8, 3): z selects projection.
// ---------------------------------------------------------------------------
__global__ __launch_bounds__(256) void k_gemm_qkv(
    const u16* __restrict__ hid,
    const u16* __restrict__ Wq, const u16* __restrict__ Wk, const u16* __restrict__ Wv,
    const float* __restrict__ bq, const float* __restrict__ bk, const float* __restrict__ bv,
    u16* __restrict__ qo, u16* __restrict__ ko, u16* __restrict__ vo)
{
    __shared__ __align__(16) u16 ldsA[2 * 128 * 32];
    __shared__ __align__(16) u16 ldsB[2 * 128 * 32];
    const u16*   W  = (blockIdx.z == 0) ? Wq : (blockIdx.z == 1) ? Wk : Wv;
    const float* bb = (blockIdx.z == 0) ? bq : (blockIdx.z == 1) ? bk : bv;
    u16*         O  = (blockIdx.z == 0) ? qo : (blockIdx.z == 1) ? ko : vo;

    const int m0 = blockIdx.x * 128, n0 = blockIdx.y * 128;
    v8f acc[8] = {};
    gemm_core_128x128(hid, W, m0, n0, D_, ldsA, ldsB, acc, threadIdx.x);

    const int wave = threadIdx.x >> 5, lane = threadIdx.x & 31;
    const int g = lane >> 4, nl = lane & 15;
    #pragma unroll
    for (int nt = 0; nt < 8; ++nt) {
        const int n = n0 + nt * 16 + nl;
        const int h = n >> 6, dk = n & 63;
        const float bias = bb[n];
        #pragma unroll
        for (int v = 0; v < 8; ++v) {
            const int m = m0 + wave * 16 + v + 8 * g;      // row in [0,2048)
            const int b = m >> 10, l = m & 1023;
            O[((size_t)(b * H_ + h) * L_ + l) * DK_ + dk] = f2bf(acc[nt][v] + bias);
        }
    }
}

// ---------------------------------------------------------------------------
// Output projection: out = ctx @ Wo^T + bo (fp32 out). grid (16, 8).
// ---------------------------------------------------------------------------
__global__ __launch_bounds__(256) void k_gemm_out(
    const u16* __restrict__ ctx, const u16* __restrict__ Wo,
    const float* __restrict__ bo, float* __restrict__ out)
{
    __shared__ __align__(16) u16 ldsA[2 * 128 * 32];
    __shared__ __align__(16) u16 ldsB[2 * 128 * 32];
    const int m0 = blockIdx.x * 128, n0 = blockIdx.y * 128;
    v8f acc[8] = {};
    gemm_core_128x128(ctx, Wo, m0, n0, D_, ldsA, ldsB, acc, threadIdx.x);

    const int wave = threadIdx.x >> 5, lane = threadIdx.x & 31;
    const int g = lane >> 4, nl = lane & 15;
    #pragma unroll
    for (int nt = 0; nt < 8; ++nt) {
        const int n = n0 + nt * 16 + nl;
        const float bias = bo[n];
        #pragma unroll
        for (int v = 0; v < 8; ++v) {
            const int m = m0 + wave * 16 + v + 8 * g;
            out[(size_t)m * D_ + n] = acc[nt][v] + bias;
        }
    }
}

// ---------------------------------------------------------------------------
// p2c_full[b,h,r,j] = sum_dk relq_s[h,r,dk] * k[b,h,j,dk]   (scale folded in)
// One wave per 16x16 tile. grid (9*64, 32), block 32.
// ---------------------------------------------------------------------------
__global__ __launch_bounds__(32) void k_p2c(
    const u16* __restrict__ relq, const u16* __restrict__ kbf, float* __restrict__ p2c)
{
    const int bh = blockIdx.y, h = bh & (H_ - 1);
    const int m0 = (blockIdx.x >> 6) * 16;        // over R (9 tiles, guarded)
    const int n0 = (blockIdx.x & 63) * 16;        // over L
    const int lane = threadIdx.x, g = lane >> 4, nl = lane & 15;

    FragBF a0, a1, b0, b1;
    const int r = min(m0 + nl, R_ - 1);
    const u16* arow = relq + (size_t)(h * R_ + r) * DK_;
    const u16* brow = kbf + ((size_t)bh * L_ + n0 + nl) * DK_;
    #pragma unroll
    for (int v = 0; v < 8; ++v) {
        const int kp = kpos(v, g);
        a0.u[v] = *reinterpret_cast<const u32*>(arow + kp);
        a1.u[v] = *reinterpret_cast<const u32*>(arow + 32 + kp);
        b0.u[v] = *reinterpret_cast<const u32*>(brow + kp);
        b1.u[v] = *reinterpret_cast<const u32*>(brow + 32 + kp);
    }
    v8f acc = {};
    acc = __builtin_amdgcn_wmma_f32_16x16x32_bf16(false, a0.bf, false, b0.bf, (short)0, acc, false, false);
    acc = __builtin_amdgcn_wmma_f32_16x16x32_bf16(false, a1.bf, false, b1.bf, (short)0, acc, false, false);
    #pragma unroll
    for (int v = 0; v < 8; ++v) {
        const int rr = m0 + v + 8 * g;
        if (rr < R_) p2c[((size_t)bh * R_ + rr) * L_ + n0 + nl] = acc[v];
    }
}

// ---------------------------------------------------------------------------
// c2p_full[b,h,i,r] = sum_dk q[b,h,i,dk] * relk_s[h,r,dk]
// grid (64*9, 32), block 32.
// ---------------------------------------------------------------------------
__global__ __launch_bounds__(32) void k_c2p(
    const u16* __restrict__ qbf, const u16* __restrict__ relk, float* __restrict__ c2p)
{
    const int bh = blockIdx.y, h = bh & (H_ - 1);
    const int m0 = (blockIdx.x / 9) * 16;         // over L (i)
    const int n0 = (blockIdx.x % 9) * 16;         // over R (guarded)
    const int lane = threadIdx.x, g = lane >> 4, nl = lane & 15;

    FragBF a0, a1, b0, b1;
    const u16* arow = qbf + ((size_t)bh * L_ + m0 + nl) * DK_;
    const int r = min(n0 + nl, R_ - 1);
    const u16* brow = relk + (size_t)(h * R_ + r) * DK_;
    #pragma unroll
    for (int v = 0; v < 8; ++v) {
        const int kp = kpos(v, g);
        a0.u[v] = *reinterpret_cast<const u32*>(arow + kp);
        a1.u[v] = *reinterpret_cast<const u32*>(arow + 32 + kp);
        b0.u[v] = *reinterpret_cast<const u32*>(brow + kp);
        b1.u[v] = *reinterpret_cast<const u32*>(brow + 32 + kp);
    }
    v8f acc = {};
    acc = __builtin_amdgcn_wmma_f32_16x16x32_bf16(false, a0.bf, false, b0.bf, (short)0, acc, false, false);
    acc = __builtin_amdgcn_wmma_f32_16x16x32_bf16(false, a1.bf, false, b1.bf, (short)0, acc, false, false);
    #pragma unroll
    for (int v = 0; v < 8; ++v) {
        const int rr = n0 + nl;
        if (rr < R_) c2p[((size_t)bh * L_ + m0 + v + 8 * g) * R_ + rr] = acc[v];
    }
}

// ---------------------------------------------------------------------------
// Flash attention with fused p2c/c2p gathers + mask + online softmax.
// grid (L/64, B*H), block 128 (4 waves); wave owns a 16-row query tile.
// K tiles: async double-buffered. V tiles: register-staged TRANSPOSED into
// LDS so the P@V B-fragment is two contiguous ds_load_b128 per operand.
// ---------------------------------------------------------------------------
__global__ __launch_bounds__(128) void k_attn(
    const u16* __restrict__ qbf, const u16* __restrict__ kbf, const u16* __restrict__ vbf,
    const float* __restrict__ p2c, const float* __restrict__ c2p,
    const int* __restrict__ pos, const unsigned char* __restrict__ mask,
    u16* __restrict__ ctx)
{
    __shared__ __align__(16) u16 ldsK[2 * 32 * 64];   // double-buffered, async
    __shared__ __align__(16) u16 ldsVT[64 * 32];      // V transposed [dk][j]
    __shared__ __align__(4)  u16 ldsP[4][16 * 32];

    const int bh = blockIdx.y, b = bh >> 4, h = bh & 15;
    const int wave = threadIdx.x >> 5, lane = threadIdx.x & 31;
    const int g = lane >> 4, nl = lane & 15;
    const int i0 = blockIdx.x * 64 + wave * 16;
    u16* ldsPw = &ldsP[wave][0];

    auto issueK = [&](int j0c, int buf) {   // 2 async instrs per wave
        u16* lk = ldsK + buf * 2048;
        for (int s = threadIdx.x; s < 256; s += 128) {
            const int row = s >> 3, c = s & 7;
            async_load_b128(lk + s * 8, kbf + ((size_t)bh * L_ + j0c + row) * DK_ + c * 8);
        }
    };
    auto loadV = [&](int j0c, uint4 vr[2]) {
        #pragma unroll
        for (int q = 0; q < 2; ++q) {
            const int s = threadIdx.x + q * 128;
            const int row = s >> 3, c = s & 7;
            vr[q] = *reinterpret_cast<const uint4*>(
                vbf + ((size_t)bh * L_ + j0c + row) * DK_ + c * 8);
        }
    };
    auto storeVT = [&](const uint4 vr[2]) {
        #pragma unroll
        for (int q = 0; q < 2; ++q) {
            const int s = threadIdx.x + q * 128;
            const int row = s >> 3, cb = (s & 7) * 8;
            u32 w;
            w = vr[q].x; ldsVT[(cb + 0) * 32 + row] = (u16)w; ldsVT[(cb + 1) * 32 + row] = (u16)(w >> 16);
            w = vr[q].y; ldsVT[(cb + 2) * 32 + row] = (u16)w; ldsVT[(cb + 3) * 32 + row] = (u16)(w >> 16);
            w = vr[q].z; ldsVT[(cb + 4) * 32 + row] = (u16)w; ldsVT[(cb + 5) * 32 + row] = (u16)(w >> 16);
            w = vr[q].w; ldsVT[(cb + 6) * 32 + row] = (u16)w; ldsVT[(cb + 7) * 32 + row] = (u16)(w >> 16);
        }
    };

    // q fragments for this wave's 16 rows (K = dk 0..31 and 32..63)
    FragBF aq0, aq1;
    {
        const u16* qrow = qbf + ((size_t)bh * L_ + i0 + nl) * DK_;
        #pragma unroll
        for (int v = 0; v < 8; ++v) {
            const int kp = kpos(v, g);
            aq0.u[v] = *reinterpret_cast<const u32*>(qrow + kp);
            aq1.u[v] = *reinterpret_cast<const u32*>(qrow + 32 + kp);
        }
    }

    v8f   acc[4] = {};
    float m_run[8], l_run[8];
    #pragma unroll
    for (int v = 0; v < 8; ++v) { m_run[v] = -3.0e38f; l_run[v] = 0.0f; }

    uint4 vreg[2];
    issueK(0, 0);
    loadV(0, vreg);

    for (int j0 = 0; j0 < L_; j0 += 32) {
        const int cur = (j0 >> 5) & 1;
        storeVT(vreg);                       // stage current V (loaded last iter)
        if (j0 + 32 < L_) {
            issueK(j0 + 32, cur ^ 1);        // stream next K tile
            loadV(j0 + 32, vreg);            // and next V into registers
            S_WAIT_ASYNC(2);                 // current K tile landed
        } else {
            S_WAIT_ASYNC(0);
        }
        __syncthreads();
        const u16* lk = ldsK + cur * 2048;

        float s0[8], s1[8];
        #pragma unroll
        for (int jsub = 0; jsub < 2; ++jsub) {
            FragBF bk0, bk1;
            const int jl = jsub * 16 + nl;
            #pragma unroll
            for (int v = 0; v < 8; ++v) {
                const int kp = kpos(v, g);
                bk0.u[v] = *reinterpret_cast<const u32*>(&lk[jl * 64 + kp]);
                bk1.u[v] = *reinterpret_cast<const u32*>(&lk[jl * 64 + 32 + kp]);
            }
            v8f s = {};
            s = __builtin_amdgcn_wmma_f32_16x16x32_bf16(false, aq0.bf, false, bk0.bf, (short)0, s, false, false);
            s = __builtin_amdgcn_wmma_f32_16x16x32_bf16(false, aq1.bf, false, bk1.bf, (short)0, s, false, false);

            const int j = j0 + jl;
            const bool mj = mask[b * L_ + j] != 0;
            const int* posT = pos + ((size_t)bh * L_ + j) * L_;   // pos[b,h,j,:]
            // prefetch next chunk's pos_indices rows (dominant HBM stream)
            __builtin_prefetch(posT + (size_t)32 * L_ + i0, 0, 1);
            __builtin_prefetch(pos + ((size_t)bh * L_ + i0 + nl) * L_ + j + 32, 0, 1);
            float* dst = jsub ? s1 : s0;
            #pragma unroll
            for (int v = 0; v < 8; ++v) {
                const int i  = i0 + v + 8 * g;
                const int r1 = pos[((size_t)bh * L_ + i) * L_ + j];  // pos[b,h,i,j]
                const int r2 = posT[i];                              // pos[b,h,j,i]
                float val = s[v] * SCALE_
                          + p2c[((size_t)bh * R_ + r1) * L_ + j]
                          + c2p[((size_t)bh * L_ + i) * R_ + r2];
                dst[v] = mj ? -1e9f : val;
            }
        }

        // online softmax over the 16x32 score tile
        #pragma unroll
        for (int v = 0; v < 8; ++v) {
            const float tmax = half16_max(fmaxf(s0[v], s1[v]));
            const float mnew = fmaxf(m_run[v], tmax);
            const float alpha = __expf(m_run[v] - mnew);
            m_run[v] = mnew;
            const float p0 = __expf(s0[v] - mnew);
            const float p1 = __expf(s1[v] - mnew);
            l_run[v] = l_run[v] * alpha + half16_sum(p0 + p1);
            #pragma unroll
            for (int t = 0; t < 4; ++t) acc[t][v] *= alpha;
            const int mrow = v + 8 * g;
            ldsPw[mrow * 32 + nl]      = f2bf(p0);
            ldsPw[mrow * 32 + 16 + nl] = f2bf(p1);
        }
        asm volatile("s_wait_dscnt 0" ::: "memory");   // P visible before re-read

        // P (16x32 bf16, A layout) @ V^T tiles (B: k=j, n=dk; contiguous reads)
        FragBF aP;
        #pragma unroll
        for (int v = 0; v < 8; ++v)
            aP.u[v] = *reinterpret_cast<const u32*>(&ldsPw[nl * 32 + kpos(v, g)]);
        #pragma unroll
        for (int t = 0; t < 4; ++t) {
            FragBF bv;
            const int dk = t * 16 + nl;
            #pragma unroll
            for (int v = 0; v < 8; ++v)
                bv.u[v] = *reinterpret_cast<const u32*>(&ldsVT[dk * 32 + kpos(v, g)]);
            acc[t] = __builtin_amdgcn_wmma_f32_16x16x32_bf16(
                false, aP.bf, false, bv.bf, (short)0, acc[t], false, false);
        }
        __syncthreads();
    }

    // normalize and emit ctx in [B, L, D] bf16 for the output GEMM
    #pragma unroll
    for (int t = 0; t < 4; ++t) {
        const int dk = t * 16 + nl;
        #pragma unroll
        for (int v = 0; v < 8; ++v) {
            const int i = i0 + v + 8 * g;
            const float o = acc[t][v] / l_run[v];
            ctx[((size_t)(b * L_) + i) * D_ + h * DK_ + dk] = f2bf(o);
        }
    }
}

// ---------------------------------------------------------------------------
// Host launcher
// ---------------------------------------------------------------------------
extern "C" void kernel_launch(void* const* d_in, const int* in_sizes, int n_in,
                              void* d_out, int out_size, void* d_ws, size_t ws_size,
                              hipStream_t stream) {
    (void)in_sizes; (void)n_in; (void)out_size; (void)ws_size;
    const float* hidden = (const float*)d_in[0];
    const float* rel_q  = (const float*)d_in[1];
    const float* rel_k  = (const float*)d_in[2];
    const float* Wq     = (const float*)d_in[3];
    const float* bq     = (const float*)d_in[4];
    const float* Wk     = (const float*)d_in[5];
    const float* bk     = (const float*)d_in[6];
    const float* Wv     = (const float*)d_in[7];
    const float* bv     = (const float*)d_in[8];
    const float* Wo     = (const float*)d_in[9];
    const float* bo     = (const float*)d_in[10];
    const int*   pos    = (const int*)d_in[11];
    const unsigned char* mask = (const unsigned char*)d_in[12];
    float* out = (float*)d_out;

    char* ws = (char*)d_ws;
    size_t off = 0;
    auto alloc = [&](size_t bytes) -> char* {
        char* p = ws + off;
        off = (off + bytes + 255) & ~(size_t)255;
        return p;
    };
    const size_t NH = (size_t)B_ * L_ * D_;        // 2,097,152
    const size_t NW = (size_t)D_ * D_;             // 1,048,576
    const size_t NR = (size_t)H_ * R_ * DK_;       // 133,120
    const size_t NQ = (size_t)B_ * H_ * L_ * DK_;  // 2,097,152
    const size_t NP = (size_t)B_ * H_ * R_ * L_;   // 4,259,840

    u16* hid_bf  = (u16*)alloc(NH * 2);
    u16* Wq_bf   = (u16*)alloc(NW * 2);
    u16* Wk_bf   = (u16*)alloc(NW * 2);
    u16* Wv_bf   = (u16*)alloc(NW * 2);
    u16* Wo_bf   = (u16*)alloc(NW * 2);
    u16* relq_bf = (u16*)alloc(NR * 2);
    u16* relk_bf = (u16*)alloc(NR * 2);
    u16* q_bf    = (u16*)alloc(NQ * 2);
    u16* k_bf    = (u16*)alloc(NQ * 2);
    u16* v_bf    = (u16*)alloc(NQ * 2);
    float* p2c   = (float*)alloc(NP * 4);
    float* c2p   = (float*)alloc(NP * 4);
    u16* ctx_bf  = (u16*)alloc(NH * 2);

    auto cvt = [&](const float* src, u16* dst, size_t n, float sc) {
        k_cvt_bf16<<<(unsigned)((n + 255) / 256), 256, 0, stream>>>(src, dst, (int)n, sc);
    };
    cvt(hidden, hid_bf, NH, 1.0f);
    cvt(Wq, Wq_bf, NW, 1.0f);
    cvt(Wk, Wk_bf, NW, 1.0f);
    cvt(Wv, Wv_bf, NW, 1.0f);
    cvt(Wo, Wo_bf, NW, 1.0f);
    cvt(rel_q, relq_bf, NR, SCALE_);   // fold score scale into rel tables
    cvt(rel_k, relk_bf, NR, SCALE_);

    k_gemm_qkv<<<dim3(16, 8, 3), 256, 0, stream>>>(
        hid_bf, Wq_bf, Wk_bf, Wv_bf, bq, bk, bv, q_bf, k_bf, v_bf);

    k_p2c<<<dim3(9 * 64, B_ * H_), 32, 0, stream>>>(relq_bf, k_bf, p2c);
    k_c2p<<<dim3(64 * 9, B_ * H_), 32, 0, stream>>>(q_bf, relk_bf, c2p);

    k_attn<<<dim3(L_ / 64, B_ * H_), 128, 0, stream>>>(
        q_bf, k_bf, v_bf, p2c, c2p, pos, mask, ctx_bf);

    k_gemm_out<<<dim3(16, 8), 256, 0, stream>>>(ctx_bf, Wo_bf, bo, out);
}
